// Attention_75677323755781
// MI455X (gfx1250) — compile-verified, matrix operand order checked by reading, not run on hardware
//
#include <hip/hip_runtime.h>
#include <hip/hip_bf16.h>
#include <math.h>

typedef __attribute__((ext_vector_type(16))) __bf16 v16bf;
typedef __attribute__((ext_vector_type(8)))  float  v8f;
typedef __attribute__((ext_vector_type(4)))  unsigned int tdm_v4u;
typedef __attribute__((ext_vector_type(8)))  int          tdm_v8i;
typedef __attribute__((ext_vector_type(4)))  int          tdm_v4i;

#define B_   4
#define T_   2048
#define D_   512
#define H_   8
#define HD_  4096
#define NROW (B_ * T_)                 // 8192
#define SCALE_ 0.04419417382415922f    // 1/sqrt(512)

// ---- WMMA fragment K-index (CDNA5 ISA 16-bit A/B layout) ----
// lane half (0: lanes 0-15, 1: lanes 16-31); VGPR i holds K={base,base+1},
// base = (i<4 ? 2i : 16+2(i-4)) + half*8  -> two contiguous 8-elem chunks:
// [half*8, half*8+8) and [16+half*8, 16+half*8+8)
__device__ __forceinline__ int ak(int i, int half) {
  return ((i < 4) ? (2 * i) : (16 + 2 * (i - 4))) + half * 8;
}

// ------------------------------------------------------------------
// f32 -> bf16 elementwise
// ------------------------------------------------------------------
__global__ __launch_bounds__(256) void cvt_f32_bf16(const float* __restrict__ s,
                                                    __bf16* __restrict__ d, int n) {
  int i = blockIdx.x * blockDim.x + threadIdx.x;
  if (i < n) d[i] = (__bf16)s[i];
}

// ------------------------------------------------------------------
// Weight convert + transpose: Wf f32 [512][4096] -> Wt bf16 [4096][512]
// ------------------------------------------------------------------
__global__ __launch_bounds__(256)
void cvt_transpose_w(const float* __restrict__ Wf, __bf16* __restrict__ Wt) {
  __shared__ __align__(16) __bf16 tile[64][72];
  const int tid = threadIdx.x;
  const int n0 = blockIdx.x * 64;   // col block in source (4096/64)
  const int d0 = blockIdx.y * 64;   // row block in source (512/64)

  for (int c = tid; c < 1024; c += 256) {       // 64 rows x 16 float4 chunks
    int r = c >> 4, g = c & 15;
    const float4 v = *reinterpret_cast<const float4*>(
        &Wf[(size_t)(d0 + r) * HD_ + n0 + g * 4]);
    __bf16* p = &tile[r][g * 4];
    p[0] = (__bf16)v.x; p[1] = (__bf16)v.y; p[2] = (__bf16)v.z; p[3] = (__bf16)v.w;
  }
  __syncthreads();
  for (int c = tid; c < 512; c += 256) {        // 64 n-rows x 8 chunks of 8 d
    int r = c >> 3, g = c & 7;
    alignas(16) __bf16 tmp[8];
#pragma unroll
    for (int j = 0; j < 8; ++j) tmp[j] = tile[g * 8 + j][r];
    *reinterpret_cast<uint4*>(&Wt[(size_t)(n0 + r) * D_ + d0 + g * 8]) =
        *reinterpret_cast<const uint4*>(tmp);
  }
}

// ------------------------------------------------------------------
// Projection GEMM: acc = X[8192,512] @ Wt^T (+bias) * outScale
//   Wt is bf16 [4096][512] (N-major, K contiguous).
// outMode 0: Out[m][n] bf16 [8192][4096]
// outMode 1: Out = Vt, bf16 [(b,h,n)][t] = [16384][2048] (transposed epilogue)
// Block 256 thr / 8 waves; tile 64(M) x 128(N); K-step 32; wave = 2x2 WMMA.
// ------------------------------------------------------------------
__global__ __launch_bounds__(256)
void proj_gemm(const __bf16* __restrict__ X, const __bf16* __restrict__ Wt,
               const float* __restrict__ bias, __bf16* __restrict__ Out,
               float outScale, int outMode) {
  __shared__ __align__(16) __bf16 As[64][40];     // 64 rows x 32 k (+pad)
  __shared__ __align__(16) __bf16 Bs[128][40];    // 128 n-rows x 32 k (+pad)
  __shared__ __align__(16) __bf16 Cs[64][136];    // transpose staging

  const int tid  = threadIdx.x;
  const int wid  = tid >> 5;
  const int lane = tid & 31;
  const int half = lane >> 4;
  const int l16  = lane & 15;
  const int rowBase = blockIdx.y * 64;
  const int colBase = blockIdx.x * 128;
  const int wm = (wid & 1) * 32;
  const int wn = (wid >> 1) * 32;

  v8f acc[2][2];
#pragma unroll
  for (int mi = 0; mi < 2; ++mi)
#pragma unroll
    for (int ni = 0; ni < 2; ++ni)
#pragma unroll
      for (int e = 0; e < 8; ++e) acc[mi][ni][e] = 0.0f;

  for (int k0 = 0; k0 < D_; k0 += 32) {
    {   // stage A 64x32: one uint4 per thread
      int r = tid >> 2, cg = tid & 3;
      const __bf16* src = &X[(size_t)(rowBase + r) * D_ + k0 + cg * 8];
      *reinterpret_cast<uint4*>(&As[r][cg * 8]) =
          *reinterpret_cast<const uint4*>(src);
      if (k0 + 32 < D_) __builtin_prefetch(src + 32, 0, 1);
    }
    {   // stage B 128x32 (from Wt, K contiguous): two uint4 per thread
      for (int c = tid; c < 512; c += 256) {
        int r = c >> 2, g = c & 3;
        const __bf16* src = &Wt[(size_t)(colBase + r) * D_ + k0 + g * 8];
        *reinterpret_cast<uint4*>(&Bs[r][g * 8]) =
            *reinterpret_cast<const uint4*>(src);
        if (k0 + 32 < D_) __builtin_prefetch(src + 32, 0, 1);
      }
    }
    __syncthreads();

    v16bf a[2], b[2];
#pragma unroll
    for (int mi = 0; mi < 2; ++mi) {
      const __bf16* pa = &As[wm + mi * 16 + l16][0];
#pragma unroll
      for (int i = 0; i < 8; ++i) {
        int k = ak(i, half);
        a[mi][2 * i]     = pa[k];
        a[mi][2 * i + 1] = pa[k + 1];
      }
    }
#pragma unroll
    for (int ni = 0; ni < 2; ++ni) {
      const __bf16* pb = &Bs[wn + ni * 16 + l16][0];
#pragma unroll
      for (int i = 0; i < 8; ++i) {
        int k = ak(i, half);
        b[ni][2 * i]     = pb[k];
        b[ni][2 * i + 1] = pb[k + 1];
      }
    }
#pragma unroll
    for (int mi = 0; mi < 2; ++mi)
#pragma unroll
      for (int ni = 0; ni < 2; ++ni)
        acc[mi][ni] = __builtin_amdgcn_wmma_f32_16x16x32_bf16(
            false, a[mi], false, b[ni], (short)0, acc[mi][ni], false, false);
    __syncthreads();
  }

  if (outMode == 0) {
    // direct epilogue: +bias, *scale, bf16 row-major
#pragma unroll
    for (int mi = 0; mi < 2; ++mi)
#pragma unroll
      for (int ni = 0; ni < 2; ++ni) {
        int n = colBase + wn + ni * 16 + l16;
        float bv = bias[n];
#pragma unroll
        for (int i = 0; i < 8; ++i) {
          int m = rowBase + wm + mi * 16 + i + half * 8;
          Out[(size_t)m * HD_ + n] = (__bf16)((acc[mi][ni][i] + bv) * outScale);
        }
      }
  } else {
    // transposed epilogue (for V): stage tile in LDS, emit Vt[(b,h,n)][t]
#pragma unroll
    for (int mi = 0; mi < 2; ++mi)
#pragma unroll
      for (int ni = 0; ni < 2; ++ni) {
        int n = wn + ni * 16 + l16;
        float bv = bias[colBase + n];
#pragma unroll
        for (int i = 0; i < 8; ++i) {
          int m = wm + mi * 16 + i + half * 8;
          Cs[m][n] = (__bf16)((acc[mi][ni][i] + bv) * outScale);
        }
      }
    __syncthreads();
    const int bb  = rowBase >> 11;        // batch (T_=2048)
    const int t0  = rowBase & (T_ - 1);
    const int h   = colBase >> 9;         // head  (D_=512)
    const int nh0 = colBase & (D_ - 1);
    for (int c = tid; c < 1024; c += 256) {   // 128 n-rows x 8 chunks of 8 t
      int r = c >> 3, g = c & 7;
      alignas(16) __bf16 tmp[8];
#pragma unroll
      for (int j = 0; j < 8; ++j) tmp[j] = Cs[g * 8 + j][r];
      *reinterpret_cast<uint4*>(
          &Out[((size_t)((bb * H_ + h) * D_ + nh0 + r)) * T_ + t0 + g * 8]) =
          *reinterpret_cast<const uint4*>(tmp);
    }
  }
}

// ------------------------------------------------------------------
// Flash attention, causal (TF-style score -= 1e9).
// Grid (T/32, H, B), 256 threads (8 waves).
// Q tile 32x512 bf16 in LDS, loaded via the Tensor Data Mover with LDS
// padding (4 DW per 256 DW row) matching Qs[32][520].
// S = 32x64 via 8 waves x one 16x16 WMMA subtile (K from global, b128).
// Online softmax in LDS; O = 32x512 f32 in regs; PV reads Vt (b128).
// ------------------------------------------------------------------
__global__ __launch_bounds__(256)
void flash_attn(const __bf16* __restrict__ Qb, const __bf16* __restrict__ Kb,
                const __bf16* __restrict__ Vt, float* __restrict__ Out) {
  __shared__ __align__(16) __bf16 Qs[32][520];
  __shared__ float  Ss[32][68];
  __shared__ __align__(16) __bf16 Ps[32][72];
  __shared__ float  rowM[32], rowL[32], rowA[32];

  const int tid  = threadIdx.x;
  const int wid  = tid >> 5;
  const int lane = tid & 31;
  const int half = lane >> 4;
  const int l16  = lane & 15;
  const int b  = blockIdx.z;
  const int h  = blockIdx.y;
  const int q0 = blockIdx.x * 32;

  const size_t baseQ  = (size_t)(b * T_ + q0) * HD_ + h * D_;
  const size_t baseK  = (size_t)(b * T_) * HD_ + h * D_;
  const size_t baseV  = (size_t)(b * H_ + h) * D_ * T_;   // Vt head base

  // ---- TDM: DMA the 32x512 bf16 Q tile into Qs with row padding ----
  if (wid == 0) {
    unsigned ldsAddr = (unsigned)(unsigned long long)(const void*)&Qs[0][0];
    unsigned long long ga = (unsigned long long)(const void*)(Qb + baseQ);
    tdm_v4u g0;
    g0[0] = 1u;                                   // count=1, user mode
    g0[1] = ldsAddr;                              // LDS byte address
    g0[2] = (unsigned)(ga & 0xffffffffu);         // global addr lo
    g0[3] = (unsigned)((ga >> 32) & 0x1ffffffu) | (2u << 30);  // hi | type=2
    tdm_v8i g1;
    g1[0] = (int)((1u << 16) |                    // data_size = 2 bytes
                  (1u << 20) |                    // pad_enable
                  (7u << 22) |                    // pad_interval = 256 DW (1 row)
                  (3u << 25));                    // pad_amount = 4 DW (8 bf16)
    g1[1] = (int)(512u << 16);                    // tensor_dim0 = 512
    g1[2] = (int)(32u << 16);                     // tensor_dim1 = 32
    g1[3] = (int)(512u << 16);                    // tile_dim0 = 512
    g1[4] = 32;                                   // tile_dim1 = 32, tile_dim2 = 0
    g1[5] = (int)4096u;                           // tensor_dim0_stride = 4096
    g1[6] = 0;
    g1[7] = 0;
    tdm_v4i gz4 = {0, 0, 0, 0};
    tdm_v8i gz8 = {0, 0, 0, 0, 0, 0, 0, 0};
    __builtin_amdgcn_tensor_load_to_lds(g0, g1, gz4, gz4, gz8, 0);
    __builtin_amdgcn_s_wait_tensorcnt(0);
  }
  if (tid < 32) { rowM[tid] = -1e30f; rowL[tid] = 0.0f; }

  const int sM = (wid & 1) * 16;          // S subtile origin
  const int sN = (wid >> 1) * 16;
  const int oM = (wid & 1) * 16;          // O region origin
  const int oN = (wid >> 1) * 128;

  v8f oacc[8];
#pragma unroll
  for (int ns = 0; ns < 8; ++ns)
#pragma unroll
    for (int e = 0; e < 8; ++e) oacc[ns][e] = 0.0f;

  __syncthreads();

  for (int kb = 0; kb <= q0 + 31; kb += 64) {
    // ---- S = Q @ K^T : one 16x16 subtile per wave ----
    v8f sacc;
#pragma unroll
    for (int e = 0; e < 8; ++e) sacc[e] = 0.0f;

    const __bf16* pkrow = &Kb[baseK + (size_t)(kb + sN + l16) * HD_];
    const __bf16* parow = &Qs[sM + l16][0];
    for (int d0 = 0; d0 < D_; d0 += 32) {
      v16bf a, bb;
#pragma unroll
      for (int i = 0; i < 8; ++i) {
        int k = ak(i, half);
        a[2 * i]      = parow[d0 + k];
        a[2 * i + 1]  = parow[d0 + k + 1];
        bb[2 * i]     = pkrow[d0 + k];
        bb[2 * i + 1] = pkrow[d0 + k + 1];
      }
      sacc = __builtin_amdgcn_wmma_f32_16x16x32_bf16(false, a, false, bb,
                                                     (short)0, sacc, false, false);
    }
#pragma unroll
    for (int i = 0; i < 8; ++i) Ss[sM + i + half * 8][sN + l16] = sacc[i];
    __syncthreads();

    // ---- online softmax: 8 lanes per row, 8 cols each ----
    {
      int r = tid >> 3, part = tid & 7;
      int qrow = q0 + r;
      float mold = rowM[r];
      float s[8];
      float mx = -1e30f;
#pragma unroll
      for (int c0 = 0; c0 < 8; ++c0) {
        int c = part * 8 + c0;
        float sv = Ss[r][c];
        if (kb + c > qrow) sv -= 1e9f;   // TF dense_attention masking
        s[c0] = sv;
        mx = fmaxf(mx, sv);
      }
#pragma unroll
      for (int off = 1; off < 8; off <<= 1) mx = fmaxf(mx, __shfl_xor(mx, off, 32));
      float mnew = fmaxf(mold, mx);
      float sum = 0.0f;
#pragma unroll
      for (int c0 = 0; c0 < 8; ++c0) {
        float p = __expf(s[c0] - mnew);
        sum += p;
        Ps[r][part * 8 + c0] = (__bf16)p;
      }
#pragma unroll
      for (int off = 1; off < 8; off <<= 1) sum += __shfl_xor(sum, off, 32);
      if (part == 0) {
        float alpha = __expf(mold - mnew);
        rowA[r] = alpha;
        rowM[r] = mnew;
        rowL[r] = rowL[r] * alpha + sum;
      }
    }
    __syncthreads();

    // ---- rescale O, then O += P @ V (V^T from global, contiguous k) ----
#pragma unroll
    for (int i = 0; i < 8; ++i) {
      float av = rowA[oM + i + half * 8];
#pragma unroll
      for (int ns = 0; ns < 8; ++ns) oacc[ns][i] *= av;
    }
    v16bf ap[2];
#pragma unroll
    for (int kh = 0; kh < 2; ++kh) {
      const __bf16* pp = &Ps[oM + l16][kh * 32];
#pragma unroll
      for (int i = 0; i < 8; ++i) {
        int k = ak(i, half);
        ap[kh][2 * i]     = pp[k];
        ap[kh][2 * i + 1] = pp[k + 1];
      }
    }
#pragma unroll
    for (int ns = 0; ns < 8; ++ns) {
      int n = oN + ns * 16 + l16;
      const __bf16* pv0 = &Vt[baseV + (size_t)n * T_ + kb];
#pragma unroll
      for (int kh = 0; kh < 2; ++kh) {
        v16bf bv;
        const __bf16* pv = pv0 + kh * 32;
#pragma unroll
        for (int i = 0; i < 8; ++i) {
          int k = ak(i, half);
          bv[2 * i]     = pv[k];
          bv[2 * i + 1] = pv[k + 1];
        }
        oacc[ns] = __builtin_amdgcn_wmma_f32_16x16x32_bf16(
            false, ap[kh], false, bv, (short)0, oacc[ns], false, false);
      }
    }
    __syncthreads();
  }

  // ---- finalize: divide by l and store f32 ----
#pragma unroll
  for (int ns = 0; ns < 8; ++ns) {
    int n = oN + ns * 16 + l16;
#pragma unroll
    for (int i = 0; i < 8; ++i) {
      int m = oM + i + half * 8;
      float l = rowL[m];
      Out[(size_t)(b * T_ + q0 + m) * HD_ + h * D_ + n] = oacc[ns][i] / l;
    }
  }
}

// ------------------------------------------------------------------
extern "C" void kernel_launch(void* const* d_in, const int* in_sizes, int n_in,
                              void* d_out, int out_size, void* d_ws, size_t ws_size,
                              hipStream_t stream) {
  const float* query = (const float*)d_in[0];
  const float* value = (const float*)d_in[1];
  const float* Wq = (const float*)d_in[2];
  const float* bq = (const float*)d_in[3];
  const float* Wk = (const float*)d_in[4];
  const float* bk = (const float*)d_in[5];
  const float* Wv = (const float*)d_in[6];
  const float* bv = (const float*)d_in[7];
  float* out = (float*)d_out;

  const size_t nQKV = (size_t)NROW * HD_;      // 33,554,432
  const size_t nX   = (size_t)NROW * D_;       //  4,194,304
  const size_t nW   = (size_t)D_ * HD_;        //  2,097,152

  __bf16* Qb  = (__bf16*)d_ws;       // [8192][4096]
  __bf16* Kb  = Qb + nQKV;           // [8192][4096]
  __bf16* Vt  = Kb + nQKV;           // [B*H*512][2048] transposed V
  __bf16* Xq  = Vt + nQKV;
  __bf16* Xv  = Xq + nX;
  __bf16* Wqt = Xv + nX;             // [4096][512]
  __bf16* Wkt = Wqt + nW;
  __bf16* Wvt = Wkt + nW;

  // 1) convert activations; convert+transpose weights
  cvt_f32_bf16<<<(int)(nX / 256), 256, 0, stream>>>(query, Xq, (int)nX);
  cvt_f32_bf16<<<(int)(nX / 256), 256, 0, stream>>>(value, Xv, (int)nX);
  dim3 wgrid(HD_ / 64, D_ / 64);
  cvt_transpose_w<<<wgrid, 256, 0, stream>>>(Wq, Wqt);
  cvt_transpose_w<<<wgrid, 256, 0, stream>>>(Wk, Wkt);
  cvt_transpose_w<<<wgrid, 256, 0, stream>>>(Wv, Wvt);

  // 2) projections (Q pre-scaled by 1/sqrt(D); V emitted transposed)
  dim3 pgrid(HD_ / 128, NROW / 64);
  proj_gemm<<<pgrid, 256, 0, stream>>>(Xq, Wqt, bq, Qb, SCALE_, 0);
  proj_gemm<<<pgrid, 256, 0, stream>>>(Xv, Wkt, bk, Kb, 1.0f, 0);
  proj_gemm<<<pgrid, 256, 0, stream>>>(Xv, Wvt, bv, Vt, 1.0f, 1);

  // 3) flash attention
  dim3 agrid(T_ / 32, H_, B_);
  flash_attn<<<agrid, 256, 0, stream>>>(Qb, Kb, Vt, out);
}
